// SparseConv3d_52767968199295
// MI455X (gfx1250) — compile-verified, matrix operand order checked by reading, not run on hardware
//
#include <hip/hip_runtime.h>
#include <hip/hip_bf16.h>

typedef _Float16 v16h __attribute__((ext_vector_type(16)));
typedef _Float16 v8h  __attribute__((ext_vector_type(8)));
typedef float    v8f  __attribute__((ext_vector_type(8)));

#define NVOX   1000000
#define CIN    32
#define COUT   32
#define KVOL   27
#define NTILES (NVOX / 16)          // 62500 exactly
#define LANE_STRIDE 24              // halfs per LDS B-fragment lane slot (48 B)

// LDS weight cache: [k][ntile][lane] -> 16 halfs (padded to 24)
// size = 27*2*32*24 halfs = 41472 halfs = 82944 bytes
#define LDSW_HALFS (KVOL * 2 * 32 * LANE_STRIDE)

__global__ __launch_bounds__(256) void sparse_conv3d_wmma(
    const float* __restrict__ feats,      // (N, 32) fp32 -- L2-resident (RT)
    const float* __restrict__ weight,     // (27, 32, 32) fp32
    const float* __restrict__ bias,       // (32,) fp32
    const int*   __restrict__ nbr,        // (27, N) int32, N = sentinel -- streamed (NT)
    float*       __restrict__ out)        // (N, 32) fp32 -- streamed (NT)
{
    __shared__ _Float16 lds_w[LDSW_HALFS];

    // ---- One-time per-workgroup: convert fp32 weights into f16 B-fragment
    //      layout in LDS. B (K=32 x N=16) as packed here (contiguous k, same as A): lane L holds
    //      column N = L%16, K-range (L<16 ? 0..15 : 16..31), 16 halfs.
    {
        const int entries = KVOL * 2 * 32;            // 1728
        for (int e = threadIdx.x; e < entries; e += blockDim.x) {
            const int k    = e >> 6;                  // /64
            const int rem  = e & 63;
            const int nt   = rem >> 5;                // 0,1 -> cout tile
            const int lane = rem & 31;
            const int kk0  = (lane & 16) ? 16 : 0;    // K half
            const int col  = nt * 16 + (lane & 15);   // cout column
            const float* wk = weight + (size_t)k * (CIN * COUT);
            _Float16* dst = &lds_w[(size_t)((k * 2 + nt) * 32 + lane) * LANE_STRIDE];
#pragma unroll
            for (int j = 0; j < 16; ++j)
                dst[j] = (_Float16)wk[(kk0 + j) * COUT + col];
        }
    }
    __syncthreads();

    const int lane   = threadIdx.x & 31;
    const int wave   = threadIdx.x >> 5;
    const int gwave  = blockIdx.x * (blockDim.x >> 5) + wave;
    const int nwaves = gridDim.x * (blockDim.x >> 5);

    const int colL = lane & 15;          // D-matrix column (and A row M)
    const int ch0  = (lane & 16);        // A K-half: input-channel offset 0/16
    const int mrow = (lane >> 4) * 8;    // D-matrix row base for this lane half

    const float bias0 = bias[colL];
    const float bias1 = bias[16 + colL];

    for (int tile = gwave; tile < NTILES; tile += nwaves) {
        const int rowbase = tile * 16;
        const int myrow   = rowbase + colL;   // voxel row this lane gathers

        v8f acc0, acc1;
#pragma unroll
        for (int r = 0; r < 8; ++r) { acc0[r] = bias0; acc1[r] = bias1; }

        // Prefetch neighbor index for k=0 (NT: idx array is streamed once,
        // keep it from evicting the L2-resident features array).
        int idx_next = __builtin_nontemporal_load(nbr + (size_t)0 * NVOX + myrow);

        for (int k = 0; k < KVOL; ++k) {
            const int idx = idx_next;
            // Issue next iteration's index load now so its latency hides
            // under this iteration's gather + 2 WMMAs (clamp keeps k=26 in
            // bounds; redundant re-load of row 26 is harmless & cached).
            const int kn = (k + 1 < KVOL) ? (k + 1) : (KVOL - 1);
            idx_next = __builtin_nontemporal_load(nbr + (size_t)kn * NVOX + myrow);

            // A fragment: 16 contiguous fp32 input channels of the neighbor
            // row (64 B), zero if neighbor is missing (sentinel == NVOX).
            float4 f[4];
            f[0] = make_float4(0.f, 0.f, 0.f, 0.f);
            f[1] = f[0]; f[2] = f[0]; f[3] = f[0];
            if (idx < NVOX) {
                const float4* src =
                    (const float4*)(feats + (size_t)idx * CIN + ch0);
                f[0] = src[0]; f[1] = src[1]; f[2] = src[2]; f[3] = src[3];
            }
            v16h a;
            const float* fv = (const float*)f;
#pragma unroll
            for (int j = 0; j < 16; ++j) a[j] = (_Float16)fv[j];

            // B fragments from LDS (16-byte aligned, 48 B lane stride)
            const _Float16* p0 =
                &lds_w[(size_t)((k * 2 + 0) * 32 + lane) * LANE_STRIDE];
            const _Float16* p1 =
                &lds_w[(size_t)((k * 2 + 1) * 32 + lane) * LANE_STRIDE];
            v8h b0l = *(const v8h*)(p0);
            v8h b0h = *(const v8h*)(p0 + 8);
            v8h b1l = *(const v8h*)(p1);
            v8h b1h = *(const v8h*)(p1 + 8);
            v16h b0 = __builtin_shufflevector(b0l, b0h,
                0,1,2,3,4,5,6,7,8,9,10,11,12,13,14,15);
            v16h b1 = __builtin_shufflevector(b1l, b1h,
                0,1,2,3,4,5,6,7,8,9,10,11,12,13,14,15);

            // D = A x B + C, fp32 accumulate
            acc0 = __builtin_amdgcn_wmma_f32_16x16x32_f16(
                false, a, false, b0, (short)0, acc0, false, false);
            acc1 = __builtin_amdgcn_wmma_f32_16x16x32_f16(
                false, a, false, b1, (short)0, acc1, false, false);
        }

        // Store D: lane holds column colL, rows mrow..mrow+7 of the tile.
        // NT stores: output is write-once, never re-read by this kernel.
        float* orow = out + (size_t)(rowbase + mrow) * COUT + colL;
#pragma unroll
        for (int r = 0; r < 8; ++r) {
            __builtin_nontemporal_store(acc0[r], orow + (size_t)r * COUT);
            __builtin_nontemporal_store(acc1[r], orow + (size_t)r * COUT + 16);
        }
    }
}

extern "C" void kernel_launch(void* const* d_in, const int* in_sizes, int n_in,
                              void* d_out, int out_size, void* d_ws, size_t ws_size,
                              hipStream_t stream) {
    const float* feats  = (const float*)d_in[0];
    const float* weight = (const float*)d_in[1];
    const float* bias   = (const float*)d_in[2];
    const int*   nbr    = (const int*)d_in[3];
    float*       out    = (float*)d_out;
    (void)in_sizes; (void)n_in; (void)out_size; (void)d_ws; (void)ws_size;

    dim3 grid(1024);
    dim3 block(256);   // 8 wave32 waves
    sparse_conv3d_wmma<<<grid, block, 0, stream>>>(feats, weight, bias, nbr, out);
}
